// MultiHeadAttention_35673998361253
// MI455X (gfx1250) — compile-verified, matrix operand order checked by reading, not run on hardware
//
#include <hip/hip_runtime.h>

// ---------------------------------------------------------------------------
// MHA block for MI455X (gfx1250, wave32, WMMA 16x16x32 f16 -> f32 acc)
// B=4, L=2048, D=512, H=8, DK=DV=64
// ---------------------------------------------------------------------------

#define B_    4
#define L_    2048
#define D_    512
#define H_    8
#define DK_   64
#define DV_   64
#define NROWS (B_ * L_)          // 8192
#define EPS_  1e-5f

#define XLD   (D_ + 8)           // padded LDS row stride (halves) for A strips
#define SLD   (L_ + 4)           // padded score row stride (floats)
#define PHLD  (L_ + 8)           // padded prob row stride (halves)

typedef __attribute__((ext_vector_type(16))) _Float16 v16h;
typedef __attribute__((ext_vector_type(8)))  _Float16 v8h;
typedef __attribute__((ext_vector_type(4)))  _Float16 v4h;
typedef __attribute__((ext_vector_type(8)))  float    v8f;
typedef __attribute__((ext_vector_type(4)))  float    v4f;

// workspace layout in _Float16 elements (total ~34 MB)
#define WS_W    0                              // 4 * 512*512 f16 weights
#define WS_QH   (4 * D_ * D_)                  // [B,H,L,DK] f16 (Q pre-scaled)
#define WS_KH   (WS_QH + B_ * H_ * L_ * DK_)   // [B,H,L,DK] f16
#define WS_VT   (WS_KH + B_ * H_ * L_ * DK_)   // [B,H,DV,L] f16 (V TRANSPOSED)
#define WS_CTX  (WS_VT + B_ * H_ * L_ * DV_)   // [NROWS, 512] f16

// ---- WMMA fragment helpers (CDNA5 wave32 layouts, CDNA5 ISA) ----

// 16 contiguous-K halves for this lane: [k0..k0+7] and [k0+16..k0+23]
__device__ inline v16h frag_from_half(const _Float16* p) {
  v8h lo = *(const v8h*)p;
  v8h hi = *(const v8h*)(p + 16);
  v16h r;
#pragma unroll
  for (int i = 0; i < 8; ++i) { r[i] = lo[i]; r[8 + i] = hi[i]; }
  return r;
}

// A-frag (16x32, row M = lane&15, K split low/high by lane<16), ld in elems.
// Also the B-frag pattern when B[k][n] = M[n][k] with M row-major.
__device__ inline v16h load_rowfrag_half(const _Float16* base, int ld) {
  int lane = threadIdx.x & 31;
  return frag_from_half(base + (size_t)(lane & 15) * ld + ((lane < 16) ? 0 : 8));
}

#define WMMA_F16(a, b, c) \
  __builtin_amdgcn_wmma_f32_16x16x32_f16(false, (a), false, (b), (short)0, (c), false, false)

// ---------------------------------------------------------------------------
// Kernel 0: convert the four 512x512 f32 weight matrices to f16 in workspace
// ---------------------------------------------------------------------------
__global__ __launch_bounds__(256) void k0_convert_weights(
    const float* __restrict__ Wq, const float* __restrict__ Wk,
    const float* __restrict__ Wv, const float* __restrict__ Wfc,
    _Float16* __restrict__ wh) {
  int idx = blockIdx.x * 256 + threadIdx.x;       // vec4 index, 0..262143
  int m  = idx >> 16;                             // which matrix
  int r4 = (idx & 65535) * 4;
  const float* src = (m == 0) ? Wq : (m == 1) ? Wk : (m == 2) ? Wv : Wfc;
  v4f x = *(const v4f*)(src + r4);
  v4h h;
#pragma unroll
  for (int i = 0; i < 4; ++i) h[i] = (_Float16)x[i];
  *(v4h*)(wh + (size_t)m * D_ * D_ + r4) = h;
}

// ---------------------------------------------------------------------------
// Kernel 1: fused QKV projection.  Y = X @ W^T + bias.
// One block per (16-row strip, projection): X strip staged once in LDS as
// f16 (padded rows -> conflict-free ds_load_b128 A-frags), 8 waves x 4 col
// tiles.  Q pre-scaled by 1/8; V stored TRANSPOSED [b,h,dv,l] for fast P@V.
// ---------------------------------------------------------------------------
__global__ __launch_bounds__(256) void k1_qkv_proj(
    const float* __restrict__ q, const float* __restrict__ k,
    const float* __restrict__ v, const _Float16* __restrict__ wh,
    const float* __restrict__ bq, const float* __restrict__ bk,
    const float* __restrict__ bv,
    _Float16* __restrict__ qh, _Float16* __restrict__ kh,
    _Float16* __restrict__ vt) {
  __shared__ _Float16 xh[16 * XLD];              // 16.25 KB staged A strip

  int rowTile = blockIdx.x;
  int proj    = blockIdx.y;
  int wave    = threadIdx.x >> 5;
  int lane    = threadIdx.x & 31;

  const float*    X    = (proj == 0) ? q : (proj == 1) ? k : v;
  const _Float16* W    = wh + (size_t)proj * D_ * D_;
  const float*    bias = (proj == 0) ? bq : (proj == 1) ? bk : bv;

  // stage + convert the 16x512 f32 strip into LDS f16 (vec4 per iteration)
  const float* Xrow = X + (size_t)rowTile * 16 * D_;
  for (int e = threadIdx.x; e < 16 * D_ / 4; e += 256) {
    int r = e >> 7, c4 = (e & 127) * 4;
    v4f x = *(const v4f*)(Xrow + (size_t)r * D_ + c4);
    v4h h;
#pragma unroll
    for (int i = 0; i < 4; ++i) h[i] = (_Float16)x[i];
    *(v4h*)(xh + r * XLD + c4) = h;
  }
  __syncthreads();

  for (int t = wave; t < 32; t += 8) {
    int o0 = t * 16;
    v8f c = {};
#pragma unroll 4
    for (int kt = 0; kt < D_; kt += 32) {
      v16h a = load_rowfrag_half(xh + kt, XLD);
      v16h b = load_rowfrag_half(W + (size_t)o0 * D_ + kt, D_);
      c = WMMA_F16(a, b, c);
    }
    int   o     = o0 + (lane & 15);
    float bval  = bias[o];
    float scale = (proj == 0) ? 0.125f : 1.0f;
    int   hd    = o >> 6, dd = o & 63;
    int   rbase = rowTile * 16 + ((lane < 16) ? 0 : 8);
#pragma unroll
    for (int r = 0; r < 8; ++r) {
      int n  = rbase + r;
      int bb = n >> 11;
      int ll = n & 2047;
      float val = (c[r] + bval) * scale;
      if (proj == 2)  // V transposed: [b,h,dv,l]
        vt[(((size_t)bb * H_ + hd) * DV_ + dd) * L_ + ll] = (_Float16)val;
      else if (proj == 0)
        qh[(((size_t)bb * H_ + hd) * L_ + ll) * DK_ + dd] = (_Float16)val;
      else
        kh[(((size_t)bb * H_ + hd) * L_ + ll) * DK_ + dd] = (_Float16)val;
    }
  }
}

// ---------------------------------------------------------------------------
// Kernel 2: attention for one (b,h,16-query strip).
// Scores->LDS f32, mask+softmax, attn->HBM once, probs->LDS f16,
// ctx = P @ V^T via WMMA (8 waves = 4 dv tiles x 2 K halves).
// grid = (128 qTiles, 32 bh), 256 thr, ~193KB dynamic LDS (of 320KB/WGP).
// ---------------------------------------------------------------------------
__global__ __launch_bounds__(256) void k2_attention(
    const _Float16* __restrict__ qh, const _Float16* __restrict__ kh,
    const _Float16* __restrict__ vt, const int* __restrict__ mask,
    float* __restrict__ attn_out, _Float16* __restrict__ ctx) {
  extern __shared__ char smem[];
  float*    s  = (float*)smem;                                  // 16*SLD f32
  _Float16* ph = (_Float16*)(smem + 16 * SLD * sizeof(float));  // 16*PHLD f16
  __shared__ float redmax[16][17];
  __shared__ float redsum[16][17];

  int bh    = blockIdx.y;
  int bIdx  = bh / H_;
  int hIdx  = bh % H_;
  int qTile = blockIdx.x;
  int wave  = threadIdx.x >> 5;
  int lane  = threadIdx.x & 31;
  int q0    = qTile * 16;

  const _Float16* Q  = qh + ((size_t)bh * L_ + q0) * DK_;
  const _Float16* K  = kh + (size_t)bh * L_ * DK_;
  const _Float16* Vt = vt + (size_t)bh * DV_ * L_;

  // prefetch this strip's mask rows (128KB) behind the WMMA score phase
  const int* mbase = mask + ((size_t)bIdx * L_ + q0) * L_;
  for (int off = threadIdx.x * 16; off < 16 * L_; off += 256 * 16)
    __builtin_prefetch(mbase + off, 0, 1);      // -> global_prefetch_b8

  // ---- Phase 1: scores (Q pre-scaled by 1/8) ----
  v16h aq0 = load_rowfrag_half(Q, DK_);         // K=0..31
  v16h aq1 = load_rowfrag_half(Q + 32, DK_);    // K=32..63
  for (int ct = wave; ct < L_ / 16; ct += 8) {
    int k0 = ct * 16;
    v8f c = {};
    c = WMMA_F16(aq0, load_rowfrag_half(K + (size_t)k0 * DK_, DK_), c);
    c = WMMA_F16(aq1, load_rowfrag_half(K + (size_t)k0 * DK_ + 32, DK_), c);
    int col = k0 + (lane & 15);
    int rb  = (lane < 16) ? 0 : 8;
#pragma unroll
    for (int r = 0; r < 8; ++r) s[(rb + r) * SLD + col] = c[r];
  }
  __syncthreads();

  // ---- Phase 2: mask + softmax, 16 threads per row ----
  int row = threadIdx.x >> 4;
  int sub = threadIdx.x & 15;
  float*     srow = s + row * SLD;
  const int* mrow = mbase + (size_t)row * L_;
  float m = -__builtin_inff();
  for (int i = sub; i < L_; i += 16) {
    float sc = srow[i];
    if (mrow[i] != 0) { sc = -__builtin_inff(); srow[i] = sc; }  // mask!=0 -> -inf
    m = fmaxf(m, sc);
  }
  redmax[row][sub] = m;
  __syncthreads();
  if (sub == 0) {
    float mm = redmax[row][0];
#pragma unroll
    for (int i = 1; i < 16; ++i) mm = fmaxf(mm, redmax[row][i]);
    redmax[row][16] = mm;
  }
  __syncthreads();
  float rowm = redmax[row][16];
  float sum = 0.f;
  for (int i = sub; i < L_; i += 16) {
    float e = __expf(srow[i] - rowm);
    srow[i] = e;
    sum += e;
  }
  redsum[row][sub] = sum;
  __syncthreads();
  if (sub == 0) {
    float ss = 0.f;
#pragma unroll
    for (int i = 0; i < 16; ++i) ss += redsum[row][i];
    redsum[row][16] = ss;
  }
  __syncthreads();
  float inv = 1.0f / redsum[row][16];
  // attn_flat layout: [h, b, q, k]
  float* arow = attn_out + (((size_t)hIdx * B_ + bIdx) * L_ + (q0 + row)) * L_;
  for (int i = sub; i < L_; i += 16) {
    float p = srow[i] * inv;
    arow[i] = p;                       // streamed once to HBM (dominant traffic)
    ph[row * PHLD + i] = (_Float16)p;
  }
  __syncthreads();

  // ---- Phase 3: ctx = P(16x2048) @ V^T; contiguous B-frags from Vt ----
  {
    int n0    = (wave & 3) * 16;          // dv tile
    int kbase = (wave >> 2) * (L_ / 2);   // K-range half
    v8f c = {};
    for (int kt = 0; kt < L_ / 2; kt += 32) {
      v16h a = load_rowfrag_half(ph + kbase + kt, PHLD);
      v16h b = load_rowfrag_half(Vt + (size_t)n0 * L_ + kbase + kt, L_);
      c = WMMA_F16(a, b, c);
    }
    int lc = lane & 15, rb = (lane < 16) ? 0 : 8;
#pragma unroll
    for (int r = 0; r < 8; ++r) s[wave * 256 + (rb + r) * 16 + lc] = c[r];
  }
  __syncthreads();
  // combine the two K-halves, write ctx tile [16 x 64] as f16
  for (int e = threadIdx.x; e < 1024; e += 256) {
    int tile = e >> 8;
    int idx  = e & 255;
    int r = idx >> 4, cc = idx & 15;
    float val = s[tile * 256 + idx] + s[(tile + 4) * 256 + idx];
    size_t n  = (size_t)bIdx * L_ + q0 + r;
    int   col = hIdx * DV_ + tile * 16 + cc;
    ctx[n * (H_ * DV_) + col] = (_Float16)val;
  }
}

// ---------------------------------------------------------------------------
// Kernel 3: out = LayerNorm(ctx @ Wfc^T + bfc + residual(q)) * gamma + beta
// ctx strip staged in LDS; grid = 512 rowTiles, 256 thr (8 waves x 4 tiles).
// ---------------------------------------------------------------------------
__global__ __launch_bounds__(256) void k3_fc_layernorm(
    const _Float16* __restrict__ ctx, const _Float16* __restrict__ wfc_h,
    const float* __restrict__ bfc, const float* __restrict__ qin,
    const float* __restrict__ gamma, const float* __restrict__ beta,
    float* __restrict__ out) {
  __shared__ _Float16 ah[16 * XLD];   // staged ctx strip
  __shared__ float xbuf[16][D_];      // 32 KB pre-norm values
  __shared__ float redA[16][17];
  __shared__ float redB[16][17];

  int rowTile = blockIdx.x;
  int wave = threadIdx.x >> 5, lane = threadIdx.x & 31;

  // stage the 16x512 f16 ctx strip (b128 per iteration)
  const _Float16* Arow = ctx + (size_t)rowTile * 16 * D_;
  for (int e = threadIdx.x; e < 16 * D_ / 8; e += 256) {
    int r = e >> 6, c8 = (e & 63) * 8;
    *(v8h*)(ah + r * XLD + c8) = *(const v8h*)(Arow + (size_t)r * D_ + c8);
  }
  __syncthreads();

  for (int t = wave; t < 32; t += 8) {
    int o0 = t * 16;
    v8f c = {};
#pragma unroll 4
    for (int kt = 0; kt < D_; kt += 32) {
      v16h a = load_rowfrag_half(ah + kt, XLD);
      v16h b = load_rowfrag_half(wfc_h + (size_t)o0 * D_ + kt, D_);
      c = WMMA_F16(a, b, c);
    }
    int o  = o0 + (lane & 15);
    int rb = (lane < 16) ? 0 : 8;
    float bval = bfc[o];
#pragma unroll
    for (int r = 0; r < 8; ++r) {
      size_t n = (size_t)rowTile * 16 + rb + r;
      xbuf[rb + r][o] = c[r] + bval + qin[n * D_ + o];
    }
  }
  __syncthreads();

  int row = threadIdx.x >> 4, sub = threadIdx.x & 15;
  float s1 = 0.f, s2 = 0.f;
  for (int i = sub; i < D_; i += 16) {
    float x = xbuf[row][i];
    s1 += x; s2 += x * x;
  }
  redA[row][sub] = s1; redB[row][sub] = s2;
  __syncthreads();
  if (sub == 0) {
    float a = 0.f, b = 0.f;
#pragma unroll
    for (int i = 0; i < 16; ++i) { a += redA[row][i]; b += redB[row][i]; }
    redA[row][16] = a * (1.0f / D_);
    redB[row][16] = b * (1.0f / D_);
  }
  __syncthreads();
  float mu   = redA[row][16];
  float var  = redB[row][16] - mu * mu;
  float rinv = rsqrtf(var + EPS_);
  size_t n = (size_t)rowTile * 16 + row;
  for (int i = sub; i < D_; i += 16)
    out[n * D_ + i] = (xbuf[row][i] - mu) * rinv * gamma[i] + beta[i];
}

// ---------------------------------------------------------------------------
extern "C" void kernel_launch(void* const* d_in, const int* in_sizes, int n_in,
                              void* d_out, int out_size, void* d_ws, size_t ws_size,
                              hipStream_t stream) {
  (void)in_sizes; (void)n_in; (void)out_size; (void)ws_size;

  const float* q     = (const float*)d_in[0];
  const float* k     = (const float*)d_in[1];
  const float* v     = (const float*)d_in[2];
  const int*   mask  = (const int*)d_in[3];
  const float* Wq    = (const float*)d_in[4];
  const float* bq    = (const float*)d_in[5];
  const float* Wk    = (const float*)d_in[6];
  const float* bk    = (const float*)d_in[7];
  const float* Wv    = (const float*)d_in[8];
  const float* bv    = (const float*)d_in[9];
  const float* Wfc   = (const float*)d_in[10];
  const float* bfc   = (const float*)d_in[11];
  const float* gamma = (const float*)d_in[12];
  const float* beta  = (const float*)d_in[13];

  _Float16* ws  = (_Float16*)d_ws;   // needs ~34 MB
  _Float16* wh  = ws + WS_W;
  _Float16* qh  = ws + WS_QH;
  _Float16* kh  = ws + WS_KH;
  _Float16* vt  = ws + WS_VT;
  _Float16* ctx = ws + WS_CTX;

  float* out  = (float*)d_out;
  float* attn = out + (size_t)B_ * L_ * D_;

  k0_convert_weights<<<1024, 256, 0, stream>>>(Wq, Wk, Wv, Wfc, wh);
  k1_qkv_proj<<<dim3(NROWS / 16, 3), 256, 0, stream>>>(
      q, k, v, wh, bq, bk, bv, qh, kh, vt);

  const int smem_bytes = 16 * SLD * 4 + 16 * PHLD * 2;   // ~193 KB
  hipFuncSetAttribute(reinterpret_cast<const void*>(k2_attention),
                      hipFuncAttributeMaxDynamicSharedMemorySize, smem_bytes);
  k2_attention<<<dim3(L_ / 16, B_ * H_), 256, smem_bytes, stream>>>(
      qh, kh, vt, mask, attn, ctx);

  k3_fc_layernorm<<<NROWS / 16, 256, 0, stream>>>(
      ctx, wh + (size_t)3 * D_ * D_, bfc, q, gamma, beta, out);
}
